// IpaMultiRigidDenoiser_84756884619594
// MI455X (gfx1250) — matrix operand order repaired and checked by reading, not compiled, not run on hardware
//
#include <hip/hip_runtime.h>
#include <hip/hip_bf16.h>
#include <math.h>

// ---------------------------------------------------------------------------
// IPA multi-rigid denoiser for MI455X (gfx1250, wave32).
// All GEMMs: v_wmma_f32_16x16x32_f16, f16 operands / f32 accum.
// Every GEMM in this net has M,N,K % 64 == 0 -> no bounds guards, no EXEC
// divergence in the WMMA loop. 64x64 macro tile, K-step 64, 8 waves,
// each wave computes 16x32 (2 accumulators, 4 WMMA per loop body).
// A/B staged via b128 global loads; B transposed in LDS so fragments are
// contiguous ds_load_b128 reads.
// ---------------------------------------------------------------------------

typedef __attribute__((ext_vector_type(16))) _Float16 v16h;
typedef __attribute__((ext_vector_type(8)))  _Float16 v8h;
typedef __attribute__((ext_vector_type(4)))  _Float16 v4h;
typedef __attribute__((ext_vector_type(8)))  float    v8f;

#define GF_ACC   1
#define GF_RELUA 2

// Model constants
#define NR_  768
#define N_   256
#define NB_  24
#define WQ_  32
#define HK_  128
#define CS_  384
#define CF_  256
#define CFP_ 64
#define CZ_  128
#define NH_  8
#define DH_  32
#define P_   8

__device__ __forceinline__ int key_start(int n) {
  int st = n * WQ_ - 48;                       // (HK-WQ)/2
  st = st < 0 ? 0 : st;
  st = st > (NR_ - HK_) ? (NR_ - HK_) : st;    // 640
  return st;
}
__device__ __forceinline__ int key_idx(int n, int k) { return key_start(n) + k; }

// ---------------------------------------------------------------------------
// WMMA GEMM: D[M,N] (+)= op(A[M,K]) * B[K,N].  M,N,K multiples of 64.
// Template flags keep the hot loop branch-free.
// ---------------------------------------------------------------------------
template <bool GATHER, bool RELUA, bool ACC>
__global__ __launch_bounds__(256) void k_gemm(
    const float* __restrict__ A, const float* __restrict__ Bw,
    float* __restrict__ D, int M, int N, int K,
    const int* __restrict__ rowIdx) {
  __shared__ _Float16 As[64 * 64];    // [row][k]   row-major
  __shared__ _Float16 Bst[64 * 64];   // [col][k]   transposed
  const int tid  = threadIdx.x;
  const int lane = tid & 31, wid = tid >> 5;
  const int brow0 = blockIdx.y * 64;
  const int bcol0 = blockIdx.x * 64;
  const int m0 = (wid >> 1) * 16;          // wave row tile
  const int n0 = (wid & 1) * 32;           // wave col tile (16x32 out)
  const int lm = lane & 15, lh = lane >> 4;

  // per-thread staging coords (4 float4 each for A and B)
  const int arow = tid >> 2;               // 0..63
  const int ac4  = tid & 3;                // float4 slot base; +4 on 2nd pass
  const int ar_g = GATHER ? rowIdx[brow0 + arow] : (brow0 + arow);

  v8f acc0 = {}, acc1 = {};
  for (int kk0 = 0; kk0 < K; kk0 += 64) {
    // ---- stage A: 64 rows x 64 k  (2 x float4 per thread per half) ----
#pragma unroll
    for (int half = 0; half < 2; ++half) {
      int c4 = ac4 + half * 4;                       // 0..7
      float4 fv = *(const float4*)(A + (size_t)ar_g * K + kk0 + 32 * half * 0 + c4 * 4 + 0);
      // note: c4 spans 0..7 -> covers k offsets 0..31; second span below
      if (RELUA) {
        fv.x = fmaxf(fv.x, 0.f); fv.y = fmaxf(fv.y, 0.f);
        fv.z = fmaxf(fv.z, 0.f); fv.w = fmaxf(fv.w, 0.f);
      }
      v4h hp; hp[0] = (_Float16)fv.x; hp[1] = (_Float16)fv.y;
      hp[2] = (_Float16)fv.z; hp[3] = (_Float16)fv.w;
      *(v4h*)&As[arow * 64 + c4 * 4] = hp;
    }
#pragma unroll
    for (int half = 0; half < 2; ++half) {
      int c4 = 8 + ac4 + half * 4;                   // 8..15 -> k 32..63
      float4 fv = *(const float4*)(A + (size_t)ar_g * K + kk0 + c4 * 4);
      if (RELUA) {
        fv.x = fmaxf(fv.x, 0.f); fv.y = fmaxf(fv.y, 0.f);
        fv.z = fmaxf(fv.z, 0.f); fv.w = fmaxf(fv.w, 0.f);
      }
      v4h hp; hp[0] = (_Float16)fv.x; hp[1] = (_Float16)fv.y;
      hp[2] = (_Float16)fv.z; hp[3] = (_Float16)fv.w;
      *(v4h*)&As[arow * 64 + c4 * 4] = hp;
    }
    // ---- stage B transposed: global rows k, cols n -> Bst[n][k] ----
#pragma unroll
    for (int it = 0; it < 4; ++it) {
      int idx = tid + it * 256;                      // 0..1023
      int rr = idx >> 4;                             // k local 0..63
      int c4 = idx & 15;                             // n/4 slot
      float4 fv = *(const float4*)(Bw + (size_t)(kk0 + rr) * N + bcol0 + c4 * 4);
      int c0 = c4 * 4;
      Bst[(c0 + 0) * 64 + rr] = (_Float16)fv.x;
      Bst[(c0 + 1) * 64 + rr] = (_Float16)fv.y;
      Bst[(c0 + 2) * 64 + rr] = (_Float16)fv.z;
      Bst[(c0 + 3) * 64 + rr] = (_Float16)fv.w;
    }
    // prefetch next K tile into L2 (global_prefetch_b8)
    if (kk0 + 64 < K) {
      __builtin_prefetch(A + (size_t)ar_g * K + kk0 + 64, 0, 1);
      __builtin_prefetch(Bw + (size_t)(kk0 + 64 + arow) * N + bcol0, 0, 1);
    }
    __syncthreads();
    // ---- 2 WMMA K-steps of 32 over the 64-wide tile ----
#pragma unroll
    for (int ks = 0; ks < 64; ks += 32) {
      // A fragment: contiguous 16 halves per lane (k = 16*lh + s), matching B
      v8h alo = *(const v8h*)&As[(m0 + lm) * 64 + ks + lh * 16];
      v8h ahi = *(const v8h*)&As[(m0 + lm) * 64 + ks + lh * 16 + 8];
      v16h af;
#pragma unroll
      for (int e = 0; e < 8; ++e) { af[e] = alo[e]; af[e + 8] = ahi[e]; }
      // B fragments: contiguous 16 halves per lane from transposed tile
      v16h bf0 = *(const v16h*)&Bst[(n0 + lm) * 64 + ks + lh * 16];
      v16h bf1 = *(const v16h*)&Bst[(n0 + 16 + lm) * 64 + ks + lh * 16];
      acc0 = __builtin_amdgcn_wmma_f32_16x16x32_f16(
          false, af, false, bf0, (short)0, acc0, false, false);
      acc1 = __builtin_amdgcn_wmma_f32_16x16x32_f16(
          false, af, false, bf1, (short)0, acc1, false, false);
    }
    __syncthreads();
  }
  // ---- epilogue ----
  const int col0 = bcol0 + n0 + lm;
#pragma unroll
  for (int j = 0; j < 8; ++j) {
    int row = brow0 + m0 + lh * 8 + j;
    size_t i0 = (size_t)row * N + col0;
    float v0 = acc0[j], v1 = acc1[j];
    if (ACC) { v0 += D[i0]; v1 += D[i0 + 16]; }
    D[i0] = v0; D[i0 + 16] = v1;
  }
}

// ------------------------- elementwise / glue kernels ----------------------

// layernorm rows: out = (x-mean)*rsqrt(var+1e-5); one block per row
__global__ void k_ln(const float* __restrict__ in, float* __restrict__ out, int C) {
  int row = blockIdx.x;
  const float* x = in + (size_t)row * C;
  float* y = out + (size_t)row * C;
  __shared__ float red[128];
  float s = 0.f;
  for (int c = threadIdx.x; c < C; c += 128) s += x[c];
  red[threadIdx.x] = s; __syncthreads();
  for (int off = 64; off > 0; off >>= 1) {
    if (threadIdx.x < off) red[threadIdx.x] += red[threadIdx.x + off];
    __syncthreads();
  }
  float mean = red[0] / C; __syncthreads();
  float v = 0.f;
  for (int c = threadIdx.x; c < C; c += 128) { float d = x[c] - mean; v += d * d; }
  red[threadIdx.x] = v; __syncthreads();
  for (int off = 64; off > 0; off >>= 1) {
    if (threadIdx.x < off) red[threadIdx.x] += red[threadIdx.x + off];
    __syncthreads();
  }
  float rs = rsqrtf(red[0] / C + 1e-5f); __syncthreads();
  for (int c = threadIdx.x; c < C; c += 128) y[c] = (x[c] - mean) * rs;
}

// index embedding: iemb[n, 0:128]=sin, [128:256]=cos
__global__ void k_iembed(const int* __restrict__ seq_idx, float* __restrict__ iemb) {
  int n = blockIdx.x, c = threadIdx.x;
  float idxf = (float)seq_idx[n];
  int kq = (c < 128) ? c : c - 128;
  float ang = idxf * 3.14159265358979f / powf(2056.0f, (2.0f * kq) / 256.0f);
  iemb[(size_t)n * 256 + c] = (c < 128) ? sinf(ang) : cosf(ang);
}

// node += one_hot(visible) @ W_seq   (row gather of W_seq)
__global__ void k_node_seq(const int* __restrict__ seq,
                           const unsigned char* __restrict__ seq_mask,
                           const unsigned char* __restrict__ snm,
                           const float* __restrict__ W_seq,
                           float* __restrict__ node) {
  int n = blockIdx.x, c = threadIdx.x;   // 384 threads
  bool tm = (!seq_mask[n]) && snm[n];
  int vis = tm ? 20 : seq[n];
  node[(size_t)n * CS_ + c] += W_seq[(size_t)vis * CS_ + c];
}

// z init: W_rel[relpos] + rbf(d(ca_i,ca_j)) @ W_rbf
__global__ void k_zinit(const float* __restrict__ trans, const int* __restrict__ seq_idx,
                        const float* __restrict__ W_rel, const float* __restrict__ W_rbf,
                        float* __restrict__ z) {
  int pij = blockIdx.x;
  int i = pij >> 8, j = pij & 255;
  __shared__ float rbf[16];
  __shared__ float dsh;
  if (threadIdx.x == 0) {
    float dx = trans[(size_t)(i * 3) * 3 + 0] - trans[(size_t)(j * 3) * 3 + 0];
    float dy = trans[(size_t)(i * 3) * 3 + 1] - trans[(size_t)(j * 3) * 3 + 1];
    float dz = trans[(size_t)(i * 3) * 3 + 2] - trans[(size_t)(j * 3) * 3 + 2];
    dsh = sqrtf(dx * dx + dy * dy + dz * dz + 1e-8f);
  }
  __syncthreads();
  if (threadIdx.x < 16) {
    float cg = 2.0f + threadIdx.x * (20.0f / 15.0f);
    float sig = 20.0f / 16.0f;
    float dd = dsh - cg;
    rbf[threadIdx.x] = expf(-dd * dd / (2.f * sig * sig));
  }
  __syncthreads();
  int rp = seq_idx[i] - seq_idx[j];
  rp = rp < -32 ? -32 : (rp > 32 ? 32 : rp); rp += 32;
  int c = threadIdx.x;  // 128 threads
  float v = W_rel[(size_t)rp * CZ_ + c];
#pragma unroll
  for (int g = 0; g < 16; ++g) v += rbf[g] * W_rbf[(size_t)g * CZ_ + c];
  z[(size_t)pij * CZ_ + c] = v;
}

// fp init: rbf(dp)@W_fp_dist + W_fp_rel[relr]; also emit gather rows for zp
__global__ void k_fpinit(const float* __restrict__ trans,
                         const float* __restrict__ W_fp_dist,
                         const float* __restrict__ W_fp_rel,
                         float* __restrict__ fp, int* __restrict__ rowidx) {
  int bq = blockIdx.x;            // ((n*32+w)*128+k)
  int k = bq & 127;
  int tw = bq >> 7;               // token = n*32+w
  int n = tw >> 5;
  int kid = key_idx(n, k);
  __shared__ float rbf[16];
  __shared__ float dsh;
  if (threadIdx.x == 0) {
    float dx = trans[(size_t)tw * 3 + 0] - trans[(size_t)kid * 3 + 0];
    float dy = trans[(size_t)tw * 3 + 1] - trans[(size_t)kid * 3 + 1];
    float dz = trans[(size_t)tw * 3 + 2] - trans[(size_t)kid * 3 + 2];
    dsh = sqrtf(dx * dx + dy * dy + dz * dz + 1e-8f);
  }
  __syncthreads();
  if (threadIdx.x < 16) {
    float cg = 2.0f + threadIdx.x * (20.0f / 15.0f);
    float sig = 20.0f / 16.0f;
    float dd = dsh - cg;
    rbf[threadIdx.x] = expf(-dd * dd / (2.f * sig * sig));
  }
  __syncthreads();
  int resq = tw / 3, resk = kid / 3;
  int rr = resq - resk;
  rr = rr < -32 ? -32 : (rr > 32 ? 32 : rr); rr += 32;
  if (threadIdx.x == 0) rowidx[bq] = resq * 256 + resk;
  int c = threadIdx.x;  // 64 threads
  float v = W_fp_rel[(size_t)rr * CFP_ + c];
#pragma unroll
  for (int g = 0; g < 16; ++g) v += rbf[g] * W_fp_dist[(size_t)g * CFP_ + c];
  fp[(size_t)bq * CFP_ + c] = v;
}

// time embedding projected by W_time -> tproj[256]
__global__ void k_tproj(const float* __restrict__ t, const float* __restrict__ W_time,
                        float* __restrict__ tproj) {
  __shared__ float te[256];
  int c = threadIdx.x;
  float tt = t[0] * 10000.f;
  int kq = (c < 128) ? c : c - 128;
  float fr = expf((float)kq * (-logf(10000.f) / 127.f));
  te[c] = (c < 128) ? sinf(tt * fr) : cosf(tt * fr);
  __syncthreads();
  float v = 0.f;
  for (int q = 0; q < 256; ++q) v += te[q] * W_time[(size_t)q * 256 + c];
  tproj[c] = v;
}

// r[t] = nf[t/3] + pos_emb[t%3] + tproj
__global__ void k_rinit(const float* __restrict__ nf, const float* __restrict__ pos_emb,
                        const float* __restrict__ tproj, float* __restrict__ r) {
  int t = blockIdx.x, c = threadIdx.x;
  r[(size_t)t * CF_ + c] = nf[(size_t)(t / 3) * CF_ + c]
                         + pos_emb[(size_t)(t % 3) * CF_ + c] + tproj[c];
}

// r[t] += sf[t/3]
__global__ void k_addgather(const float* __restrict__ sf, float* __restrict__ r) {
  int t = blockIdx.x, c = threadIdx.x;
  r[(size_t)t * CF_ + c] += sf[(size_t)(t / 3) * CF_ + c];
}

// rotate local points to global: pg = R*pl + trans
__global__ void k_rotpts(const float* __restrict__ rot, const float* __restrict__ trans,
                         const float* __restrict__ pl, float* __restrict__ pg) {
  int t = blockIdx.x;      // 0..767
  int hp = threadIdx.x;    // 0..63 (h*8+p)
  const float* R = rot + (size_t)t * 9;
  const float* v = pl + (size_t)t * 192 + hp * 3;
  float x = v[0], y = v[1], z = v[2];
  float* o = pg + (size_t)t * 192 + hp * 3;
  o[0] = R[0] * x + R[1] * y + R[2] * z + trans[(size_t)t * 3 + 0];
  o[1] = R[3] * x + R[4] * y + R[5] * z + trans[(size_t)t * 3 + 1];
  o[2] = R[6] * x + R[7] * y + R[8] * z + trans[(size_t)t * 3 + 2];
}

// logits[n,w,k,h] = wL*(q.k/sqrt(DH) + fp.Wbp - 0.5*wC*softplus(hw)*d2)
__global__ __launch_bounds__(256) void k_logits(
    const float* __restrict__ q, const float* __restrict__ kf,
    const float* __restrict__ qpg, const float* __restrict__ kpg,
    const float* __restrict__ fp, const float* __restrict__ Wbp,
    const float* __restrict__ head_w, float* __restrict__ logits) {
  int n = blockIdx.x, h = blockIdx.y;
  __shared__ float qs[32 * 32];
  __shared__ float ks[128 * 32];
  __shared__ float qps[32 * 24];
  __shared__ float kps[128 * 24];
  __shared__ float wbp[64];
  int tid = threadIdx.x;
  for (int u = tid; u < 1024; u += 256) {
    int w = u >> 5, d = u & 31;
    qs[u] = q[(size_t)(n * 32 + w) * 256 + h * 32 + d];
  }
  for (int u = tid; u < 4096; u += 256) {
    int k = u >> 5, d = u & 31;
    ks[u] = kf[(size_t)key_idx(n, k) * 256 + h * 32 + d];
  }
  for (int u = tid; u < 768; u += 256) {
    int w = u / 24, e = u % 24;
    qps[u] = qpg[(size_t)(n * 32 + w) * 192 + h * 24 + e];
  }
  for (int u = tid; u < 3072; u += 256) {
    int k = u / 24, e = u % 24;
    kps[u] = kpg[(size_t)key_idx(n, k) * 192 + h * 24 + e];
  }
  if (tid < 64) wbp[tid] = Wbp[(size_t)tid * 8 + h];
  __syncthreads();
  float gamma = log1pf(expf(head_w[h]));
  const float wC = 0.16666667f;      // sqrt(2/(9*8))
  const float wL = 0.57735027f;      // sqrt(1/3)
  const float isq = 0.17677670f;     // 1/sqrt(32)
  for (int it = 0; it < 16; ++it) {
    int u = tid + it * 256;          // 0..4095
    int w = u >> 7, k = u & 127;
    float qk = 0.f;
    for (int d = 0; d < 32; ++d) qk += qs[w * 32 + d] * ks[k * 32 + d];
    float bias = 0.f;
    const float* fpr = fp + ((size_t)((n * 32 + w) * 128) + k) * 64;
    for (int c = 0; c < 64; ++c) bias += fpr[c] * wbp[c];
    float d2 = 0.f;
    for (int e = 0; e < 24; ++e) { float df = qps[w * 24 + e] - kps[k * 24 + e]; d2 += df * df; }
    logits[((size_t)(n * 32 + w) * 128 + k) * 8 + h] =
        wL * (qk * isq + bias - 0.5f * wC * gamma * d2);
  }
}

// softmax over k (128) for each (token,h)
__global__ void k_softmax(float* __restrict__ logits) {
  int rb = blockIdx.x;            // (token)*8 + h
  int twh = rb >> 3, h = rb & 7;
  float* base = logits + (size_t)twh * 128 * 8 + h;
  __shared__ float red[128];
  int k = threadIdx.x;
  float v = base[(size_t)k * 8];
  red[k] = v; __syncthreads();
  for (int off = 64; off > 0; off >>= 1) {
    if (k < off) red[k] = fmaxf(red[k], red[k + off]);
    __syncthreads();
  }
  float mx = red[0]; __syncthreads();
  float e = expf(v - mx);
  red[k] = e; __syncthreads();
  for (int off = 64; off > 0; off >>= 1) {
    if (k < off) red[k] += red[k + off];
    __syncthreads();
  }
  base[(size_t)k * 8] = e / red[0];
}

// attention apply: build cat = [o | ol | onorm | opair] (1024 per token)
__global__ __launch_bounds__(256) void k_attn_apply(
    const float* __restrict__ a, const float* __restrict__ vf,
    const float* __restrict__ kpg, const float* __restrict__ fp,
    const float* __restrict__ rot, const float* __restrict__ trans,
    float* __restrict__ cat) {
  int t = blockIdx.x;          // token = n*32+w
  int n = t >> 5;
  __shared__ float as_[128 * 8];
  __shared__ float Rq[9];
  __shared__ float tq[3];
  int tid = threadIdx.x;
  for (int u = tid; u < 1024; u += 256) as_[u] = a[(size_t)t * 1024 + u];
  if (tid < 9) Rq[tid] = rot[(size_t)t * 9 + tid];
  if (tid < 3) tq[tid] = trans[(size_t)t * 3 + tid];
  __syncthreads();
  float* out = cat + (size_t)t * 1024;
  for (int it = 0; it < 4; ++it) {
    int u = tid + it * 256;
    if (u < 256) {                                  // o[h,d]
      int h = u >> 5, d = u & 31;
      float v = 0.f;
      for (int k = 0; k < 128; ++k)
        v += as_[k * 8 + h] * vf[(size_t)key_idx(n, k) * 256 + h * 32 + d];
      out[u] = v;
    } else if (u < 320) {                           // ol[h,p,:] + onorm[h,p]
      int e = u - 256, h = e >> 3, p = e & 7;
      float og0 = 0.f, og1 = 0.f, og2 = 0.f;
      for (int k = 0; k < 128; ++k) {
        float ak = as_[k * 8 + h];
        const float* kp = kpg + (size_t)key_idx(n, k) * 192 + h * 24 + p * 3;
        og0 += ak * kp[0]; og1 += ak * kp[1]; og2 += ak * kp[2];
      }
      float dx = og0 - tq[0], dy = og1 - tq[1], dz = og2 - tq[2];
      float ol0 = Rq[0] * dx + Rq[3] * dy + Rq[6] * dz;
      float ol1 = Rq[1] * dx + Rq[4] * dy + Rq[7] * dz;
      float ol2 = Rq[2] * dx + Rq[5] * dy + Rq[8] * dz;
      out[256 + h * 24 + p * 3 + 0] = ol0;
      out[256 + h * 24 + p * 3 + 1] = ol1;
      out[256 + h * 24 + p * 3 + 2] = ol2;
      out[448 + h * 8 + p] = sqrtf(ol0 * ol0 + ol1 * ol1 + ol2 * ol2 + 1e-8f);
    } else if (u < 832) {                           // opair[h,c]
      int e = u - 320, h = e >> 6, c = e & 63;
      float v = 0.f;
      const float* fpr = fp + (size_t)t * 128 * 64 + c;
      for (int k = 0; k < 128; ++k) v += as_[k * 8 + h] * fpr[(size_t)k * 64];
      out[512 + h * 64 + c] = v;
    }
  }
}

// rmean[n] = mean over R of r
__global__ void k_rmean(const float* __restrict__ r, float* __restrict__ rm) {
  int n = blockIdx.x, c = threadIdx.x;
  rm[(size_t)n * CF_ + c] =
      (r[(size_t)(3 * n) * CF_ + c] + r[(size_t)(3 * n + 1) * CF_ + c] +
       r[(size_t)(3 * n + 2) * CF_ + c]) * (1.f / 3.f);
}

__global__ void k_copy(const float* __restrict__ in, float* __restrict__ out, int n) {
  int i = blockIdx.x * blockDim.x + threadIdx.x;
  if (i < n) out[i] = in[i];
}

// ---------------------------------------------------------------------------
static inline void launch_gemm(const float* A, const float* Bw, float* D,
                               int M, int N, int K, int flags, const int* rowIdx,
                               hipStream_t s) {
  dim3 g(N / 64, M / 64);
  const bool rel = (flags & GF_RELUA) != 0;
  const bool acc = (flags & GF_ACC) != 0;
  if (rowIdx) {
    if (rel) {
      if (acc) k_gemm<true, true, true><<<g, 256, 0, s>>>(A, Bw, D, M, N, K, rowIdx);
      else     k_gemm<true, true, false><<<g, 256, 0, s>>>(A, Bw, D, M, N, K, rowIdx);
    } else {
      if (acc) k_gemm<true, false, true><<<g, 256, 0, s>>>(A, Bw, D, M, N, K, rowIdx);
      else     k_gemm<true, false, false><<<g, 256, 0, s>>>(A, Bw, D, M, N, K, rowIdx);
    }
  } else {
    if (rel) {
      if (acc) k_gemm<false, true, true><<<g, 256, 0, s>>>(A, Bw, D, M, N, K, rowIdx);
      else     k_gemm<false, true, false><<<g, 256, 0, s>>>(A, Bw, D, M, N, K, rowIdx);
    } else {
      if (acc) k_gemm<false, false, true><<<g, 256, 0, s>>>(A, Bw, D, M, N, K, rowIdx);
      else     k_gemm<false, false, false><<<g, 256, 0, s>>>(A, Bw, D, M, N, K, rowIdx);
    }
  }
}

extern "C" void kernel_launch(void* const* d_in, const int* in_sizes, int n_in,
                              void* d_out, int out_size, void* d_ws, size_t ws_size,
                              hipStream_t stream) {
  (void)in_sizes; (void)n_in; (void)out_size; (void)ws_size;
  const float* t_in      = (const float*)d_in[0];
  const float* trans     = (const float*)d_in[1];
  const float* rot       = (const float*)d_in[2];
  const int*   seq_idx   = (const int*)d_in[3];
  const int*   seq       = (const int*)d_in[4];
  const unsigned char* seq_mask = (const unsigned char*)d_in[5];
  const unsigned char* snm      = (const unsigned char*)d_in[6];
  const float* W_seq     = (const float*)d_in[7];
  const float* W_node    = (const float*)d_in[8];
  const float* W_time    = (const float*)d_in[9];
  const float* W_frame   = (const float*)d_in[10];
  const float* pos_emb   = (const float*)d_in[11];
  const float* W_rel     = (const float*)d_in[12];
  const float* W_rbf     = (const float*)d_in[13];
  const float* W_eb      = (const float*)d_in[14];
  const float* W_fp_dist = (const float*)d_in[15];
  const float* W_fp_rel  = (const float*)d_in[16];
  const float* W_z2fp    = (const float*)d_in[17];
  const float* Wq        = (const float*)d_in[18];
  const float* Wk        = (const float*)d_in[19];
  const float* Wv        = (const float*)d_in[20];
  const float* Wqp       = (const float*)d_in[21];
  const float* Wkp       = (const float*)d_in[22];
  const float* Wbp       = (const float*)d_in[23];
  const float* head_w    = (const float*)d_in[24];
  const float* Wo        = (const float*)d_in[25];
  const float* Ws2f      = (const float*)d_in[26];
  const float* Wf1       = (const float*)d_in[27];
  const float* Wf2       = (const float*)d_in[28];
  const float* Wfp1      = (const float*)d_in[29];
  const float* Wfp2      = (const float*)d_in[30];
  const float* Wr2s      = (const float*)d_in[31];
  const float* Ws1       = (const float*)d_in[32];
  const float* Ws2       = (const float*)d_in[33];

  float* sOut = (float*)d_out;   // s lives in d_out (256 x 384)

  // ---- workspace bump allocation (floats) ----
  float* W = (float*)d_ws;
  size_t off = 0;
  auto alloc = [&](size_t n) { float* p = W + off; off += n; return p; };
  float* iemb  = alloc(256 * 256);
  float* node  = alloc(256 * 384);
  float* z     = alloc((size_t)65536 * 128);   // dead after zp gather-gemm
  float* zl    = alloc((size_t)65536 * 128);   // dead after z-MLP
  float* fp    = zl;                           // alias (98304 x 64 fits)
  float* fpl   = z;                            // alias (98304 x 64 fits)
  float* fpt   = alloc((size_t)98304 * 64);
  float* r     = alloc(768 * 256);
  float* x     = alloc(768 * 256);
  float* q     = alloc(768 * 256);
  float* kfb   = alloc(768 * 256);
  float* vfb   = alloc(768 * 256);
  float* qpl   = alloc(768 * 192);
  float* kpl   = alloc(768 * 192);
  float* qpg   = alloc(768 * 192);
  float* kpg   = alloc(768 * 192);
  float* logit = alloc((size_t)768 * 128 * 8);
  float* cat   = alloc((size_t)768 * 1024);
  float* t1    = alloc(768 * 512);
  float* sf    = alloc(256 * 256);
  float* rmean = alloc(256 * 256);
  float* sl    = alloc(256 * 384);
  float* t2    = alloc(256 * 768);
  float* tproj = alloc(256);
  int*   rowidx = (int*)alloc(98304);

  // ---- phase 0: node / pair embeddings ----
  k_iembed<<<256, 256, 0, stream>>>(seq_idx, iemb);
  launch_gemm(iemb, W_node, node, 256, 384, 256, 0, nullptr, stream);
  k_node_seq<<<256, 384, 0, stream>>>(seq, seq_mask, snm, W_seq, node);

  k_zinit<<<65536, 128, 0, stream>>>(trans, seq_idx, W_rel, W_rbf, z);
  for (int it = 0; it < 2; ++it) {
    k_ln<<<65536, 128, 0, stream>>>(z, zl, 128);
    launch_gemm(zl, W_eb + (size_t)it * 128 * 128, z, 65536, 128, 128,
                GF_ACC | GF_RELUA, nullptr, stream);
  }

  k_fpinit<<<98304, 64, 0, stream>>>(trans, W_fp_dist, W_fp_rel, fp, rowidx);
  launch_gemm(z, W_z2fp, fp, 98304, 64, 128, GF_ACC, rowidx, stream);  // zp @ W_z2fp

  launch_gemm(node, W_frame, sf, 256, 256, 384, 0, nullptr, stream);   // nf in sf
  k_tproj<<<1, 256, 0, stream>>>(t_in, W_time, tproj);
  k_rinit<<<768, 256, 0, stream>>>(sf, pos_emb, tproj, r);

  k_copy<<<(98304 + 255) / 256, 256, 0, stream>>>(node, sOut, 98304);  // s = node

  // ---- 3 IPA blocks ----
  for (int i = 0; i < 3; ++i) {
    const float* Wq_i   = Wq   + (size_t)i * 256 * 256;
    const float* Wk_i   = Wk   + (size_t)i * 256 * 256;
    const float* Wv_i   = Wv   + (size_t)i * 256 * 256;
    const float* Wqp_i  = Wqp  + (size_t)i * 256 * 192;
    const float* Wkp_i  = Wkp  + (size_t)i * 256 * 192;
    const float* Wbp_i  = Wbp  + (size_t)i * 64 * 8;
    const float* Wo_i   = Wo   + (size_t)i * 1024 * 256;
    const float* Ws2f_i = Ws2f + (size_t)i * 384 * 256;
    const float* Wf1_i  = Wf1  + (size_t)i * 256 * 512;
    const float* Wf2_i  = Wf2  + (size_t)i * 512 * 256;
    const float* Wfp1_i = Wfp1 + (size_t)i * 64 * 64;
    const float* Wfp2_i = Wfp2 + (size_t)i * 64 * 64;
    const float* Wr2s_i = Wr2s + (size_t)i * 256 * 384;
    const float* Ws1_i  = Ws1  + (size_t)i * 384 * 768;
    const float* Ws2_i  = Ws2  + (size_t)i * 768 * 384;
    const float* hw_i   = head_w + (size_t)i * 8;

    // fp = fp + relu(ln(fp) @ Wfp1) @ Wfp2
    k_ln<<<98304, 128, 0, stream>>>(fp, fpl, 64);
    launch_gemm(fpl, Wfp1_i, fpt, 98304, 64, 64, 0, nullptr, stream);
    launch_gemm(fpt, Wfp2_i, fp, 98304, 64, 64, GF_ACC | GF_RELUA, nullptr, stream);

    // r = r + (s @ Ws2f)[:, R2RES]
    launch_gemm(sOut, Ws2f_i, sf, 256, 256, 384, 0, nullptr, stream);
    k_addgather<<<768, 256, 0, stream>>>(sf, r);

    // x = ln(r); projections
    k_ln<<<768, 128, 0, stream>>>(r, x, 256);
    launch_gemm(x, Wq_i,  q,   768, 256, 256, 0, nullptr, stream);
    launch_gemm(x, Wk_i,  kfb, 768, 256, 256, 0, nullptr, stream);
    launch_gemm(x, Wv_i,  vfb, 768, 256, 256, 0, nullptr, stream);
    launch_gemm(x, Wqp_i, qpl, 768, 192, 256, 0, nullptr, stream);
    launch_gemm(x, Wkp_i, kpl, 768, 192, 256, 0, nullptr, stream);
    k_rotpts<<<768, 64, 0, stream>>>(rot, trans, qpl, qpg);
    k_rotpts<<<768, 64, 0, stream>>>(rot, trans, kpl, kpg);

    // attention
    k_logits<<<dim3(24, 8), 256, 0, stream>>>(q, kfb, qpg, kpg, fp, Wbp_i, hw_i, logit);
    k_softmax<<<6144, 128, 0, stream>>>(logit);
    k_attn_apply<<<768, 256, 0, stream>>>(logit, vfb, kpg, fp, rot, trans, cat);
    launch_gemm(cat, Wo_i, r, 768, 256, 1024, GF_ACC, nullptr, stream);

    // r = r + relu(ln(r) @ Wf1) @ Wf2
    k_ln<<<768, 128, 0, stream>>>(r, x, 256);
    launch_gemm(x, Wf1_i, t1, 768, 512, 256, 0, nullptr, stream);
    launch_gemm(t1, Wf2_i, r, 768, 256, 512, GF_ACC | GF_RELUA, nullptr, stream);

    // s = s + mean_R(r) @ Wr2s;  s = s + relu(ln(s) @ Ws1) @ Ws2
    k_rmean<<<256, 256, 0, stream>>>(r, rmean);
    launch_gemm(rmean, Wr2s_i, sOut, 256, 384, 256, GF_ACC, nullptr, stream);
    k_ln<<<256, 128, 0, stream>>>(sOut, sl, 384);
    launch_gemm(sl, Ws1_i, t2, 256, 768, 384, 0, nullptr, stream);
    launch_gemm(t2, Ws2_i, sOut, 256, 384, 768, GF_ACC | GF_RELUA, nullptr, stream);
  }
}
